// Evoformer3bEncoderLayer_38019050504728
// MI455X (gfx1250) — matrix operand order repaired and checked by reading, not compiled, not run on hardware
//
#include <hip/hip_runtime.h>

#define DEV __device__ __forceinline__

typedef __attribute__((ext_vector_type(16))) __bf16 v16bf;
typedef __attribute__((ext_vector_type(8)))  __bf16 v8bf;
typedef __attribute__((ext_vector_type(8)))  float  v8f;

#define B_   4
#define N_   512
#define E_   768
#define P_   64
#define PH_  32
#define H_   8
#define D_   96
#define FFN_ 3072
#define BN_  (B_*N_)
#define BNN_ (B_*N_*N_)

// ---------- scalar helpers ----------
DEV __bf16 f2bf(float f) {
  union { float f; unsigned u; } a; a.f = f;
  unsigned r = (a.u + 0x7fffu + ((a.u >> 16) & 1u)) >> 16;
  union { unsigned short s; __bf16 b; } o; o.s = (unsigned short)r; return o.b;
}
DEV float bf2f(__bf16 b) {
  union { unsigned short s; __bf16 b; } i; i.b = b;
  union { unsigned u; float f; } o; o.u = ((unsigned)i.s) << 16; return o.f;
}
DEV float geluf(float x) { return 0.5f * x * (1.f + erff(x * 0.70710678118f)); }
DEV float sigmf(float x) { return 1.f / (1.f + __expf(-x)); }
DEV v8f vzero() { v8f z = {0.f, 0.f, 0.f, 0.f, 0.f, 0.f, 0.f, 0.f}; return z; }

// ---------- WMMA fragment loaders (ISA 7.12.2, wave32) ----------
// A 16x32 bf16: lanes 0-15 -> M=lane, K=0..7 & 16..23 ; lanes 16-31 -> K=+8
DEV v16bf load_a(const __bf16* p, long ld, int lane) {
  int r = lane & 15, ko = (lane >> 4) << 3;
  const __bf16* q = p + (long)r * ld + ko;
  v16bf f;
  ((v8bf*)&f)[0] = *(const v8bf*)q;
  ((v8bf*)&f)[1] = *(const v8bf*)(q + 16);
  return f;
}
// B 32x16 bf16: lanes 0-15 -> N=lane, K=0..7 & 16..23 ; lanes 16-31 -> K=+8
DEV v16bf load_b(const __bf16* p, long ld, int lane) {
  int n = lane & 15, kb = (lane >> 4) << 3;
  const __bf16* q = p + (long)n * ld + kb;
  v16bf f;
  ((v8bf*)&f)[0] = *(const v8bf*)q;
  ((v8bf*)&f)[1] = *(const v8bf*)(q + 16);
  return f;
}
DEV v16bf load_b_strided(const __bf16* p, long nstride, long kstride, int lane) {
  int n = lane & 15, kb = (lane >> 4) << 3;
  v16bf f;
#pragma unroll
  for (int i = 0; i < 16; i++) f[i] = p[(long)n * nstride + (long)(kb + (i < 8 ? i : i + 8)) * kstride];
  return f;
}
DEV v8f wmma_bf(v16bf a, v16bf b, v8f c) {
  return __builtin_amdgcn_wmma_f32_16x16x32_bf16(false, a, false, b, (short)0, c, false, false);
}

// ---------- fp32 -> bf16 convert ----------
__global__ void cvt_f2bf(const float* __restrict__ in, __bf16* __restrict__ out, long n) {
  long i = (long)blockIdx.x * 256 + threadIdx.x;
  if (i < n) out[i] = f2bf(in[i]);
}

// ---------- generic row LayerNorm (C multiple of 64, C<=768), fp32 in -> bf16 out ----------
__global__ void ln_rows(const float* __restrict__ in, const float* __restrict__ g,
                        const float* __restrict__ b, __bf16* __restrict__ out, int C) {
  long row = blockIdx.x; int tid = threadIdx.x;
  const float* p = in + row * (long)C;
  float vals[12]; int ne = C >> 6;
  float s = 0.f;
  for (int e = 0; e < ne; e++) { vals[e] = p[tid + e * 64]; s += vals[e]; }
  __shared__ float red[64];
  red[tid] = s; __syncthreads();
  for (int off = 32; off; off >>= 1) { if (tid < off) red[tid] += red[tid + off]; __syncthreads(); }
  float mu = red[0] / (float)C; __syncthreads();
  float q = 0.f;
  for (int e = 0; e < ne; e++) { vals[e] -= mu; q += vals[e] * vals[e]; }
  red[tid] = q; __syncthreads();
  for (int off = 32; off; off >>= 1) { if (tid < off) red[tid] += red[tid + off]; __syncthreads(); }
  float inv = rsqrtf(red[0] / (float)C + 1e-5f);
  for (int e = 0; e < ne; e++) {
    int c = tid + e * 64;
    out[row * (long)C + c] = f2bf(vals[e] * inv * g[c] + b[c]);
  }
}

// ---------- fused pair LayerNorm + bias einsum('bijp,hp->bhij') + bb ----------
__global__ void pair_bias(const float* __restrict__ pair, const float* __restrict__ g,
                          const float* __restrict__ be, const float* __restrict__ wb,
                          const float* __restrict__ bb, float* __restrict__ bias) {
  long row = (long)blockIdx.x * 8 + (threadIdx.x >> 5);
  int lane = threadIdx.x & 31;
  const float* pr = pair + row * 64;
  float v0 = pr[lane], v1 = pr[lane + 32];
  float s = v0 + v1;
  for (int m = 16; m; m >>= 1) s += __shfl_xor(s, m, 32);
  float mu = s * (1.f / 64.f);
  float d0 = v0 - mu, d1 = v1 - mu;
  float q = d0 * d0 + d1 * d1;
  for (int m = 16; m; m >>= 1) q += __shfl_xor(q, m, 32);
  float inv = rsqrtf(q * (1.f / 64.f) + 1e-5f);
  float n0 = d0 * inv * g[lane] + be[lane];
  float n1 = d1 * inv * g[lane + 32] + be[lane + 32];
  long bidx = row >> 18; long ij = row & (long)(N_ * N_ - 1);
  for (int hh = 0; hh < H_; hh++) {
    float t = n0 * wb[hh * 64 + lane] + n1 * wb[hh * 64 + lane + 32];
    for (int m = 16; m; m >>= 1) t += __shfl_xor(t, m, 32);
    if (lane == 0) bias[((bidx * H_ + hh) << 18) + ij] = t + bb[hh];
  }
}

// ---------- generic WMMA GEMM, 16x64 per wave: out = act(A @ W^T + bias) ----------
#define ACT_NONE 0
#define ACT_GELU 1
#define ACT_SIG  2
template <int ACT, bool RESID>
__global__ __launch_bounds__(32) void gemm_rm(
    const __bf16* __restrict__ A, int lda,
    const __bf16* __restrict__ W, int ldw,
    const float* __restrict__ bias,
    __bf16* __restrict__ outb, float* __restrict__ outf,
    const float* __restrict__ resid, const float* __restrict__ rowscale,
    int Ncols, int K, int ldo) {
  int ng = Ncols >> 6;                 // 64-wide column groups
  int tile = blockIdx.x;
  int nt = tile % ng, mt = tile / ng;
  int lane = threadIdx.x;
  const __bf16* Ab = A + (long)mt * 16 * lda;
  const __bf16* Wb = W + (long)nt * 64 * ldw;
  v8f acc[4];
#pragma unroll
  for (int c = 0; c < 4; c++) acc[c] = vzero();
  for (int k = 0; k < K; k += 32) {
    v16bf a = load_a(Ab + k, lda, lane);
#pragma unroll
    for (int c = 0; c < 4; c++)
      acc[c] = wmma_bf(a, load_b(Wb + (long)c * 16 * ldw + k, ldw, lane), acc[c]);
  }
  int m0 = mt * 16 + ((lane >> 4) << 3);
#pragma unroll
  for (int c = 0; c < 4; c++) {
    int n = nt * 64 + c * 16 + (lane & 15);
    float bv = bias ? bias[n] : 0.f;
#pragma unroll
    for (int v = 0; v < 8; v++) {
      int m = m0 + v;
      float r = acc[c][v] + bv;
      if (ACT == ACT_GELU) r = geluf(r);
      if (ACT == ACT_SIG)  r = sigmf(r);
      if (rowscale) r *= rowscale[m];
      long idx = (long)m * ldo + n;
      if (RESID) outf[idx] = resid[idx] + r;
      else       outb[idx] = f2bf(r);
    }
  }
}

// ---------- qkv projection, 16x64 per wave, head-layout epilogue ----------
template <int MODE>   // 0:q scaled, 1:k, 2:v transposed
__global__ __launch_bounds__(32) void gemm_qkv(const __bf16* __restrict__ A,
                                               const __bf16* __restrict__ W,
                                               __bf16* __restrict__ out, float scale) {
  int ng = E_ >> 6;   // 12
  int tile = blockIdx.x;
  int nt = tile % ng, mt = tile / ng;
  int lane = threadIdx.x;
  const __bf16* Ab = A + (long)mt * 16 * E_;
  const __bf16* Wb = W + (long)nt * 64 * E_;
  v8f acc[4];
#pragma unroll
  for (int c = 0; c < 4; c++) acc[c] = vzero();
  for (int k = 0; k < E_; k += 32) {
    v16bf a = load_a(Ab + k, E_, lane);
#pragma unroll
    for (int c = 0; c < 4; c++)
      acc[c] = wmma_bf(a, load_b(Wb + (long)c * 16 * E_ + k, E_, lane), acc[c]);
  }
  int m0 = mt * 16 + ((lane >> 4) << 3);
#pragma unroll
  for (int c = 0; c < 4; c++) {
    int e = nt * 64 + c * 16 + (lane & 15);
    int hh = e / D_, d = e % D_;
#pragma unroll
    for (int v = 0; v < 8; v++) {
      int m = m0 + v; int b = m >> 9, nn = m & (N_ - 1);
      float r = acc[c][v] * scale;
      long idx;
      if (MODE == 2) idx = (((long)(b * H_ + hh)) * D_ + d) * N_ + nn;   // v^T : [B,H,D,N]
      else           idx = (((long)(b * H_ + hh)) * N_ + nn) * D_ + d;   // q,k : [B,H,N,D]
      out[idx] = f2bf(r);
    }
  }
}

// ---------- attention logits: S = q k^T + mask + bias (16x64 per wave) ----------
__global__ __launch_bounds__(32) void attn_logits(const __bf16* __restrict__ q,
                                                  const __bf16* __restrict__ k,
                                                  const float* __restrict__ mask,
                                                  const float* __restrict__ bias,
                                                  float* __restrict__ S) {
  int x = blockIdx.x; int jt4 = x & 7; x >>= 3; int it = x & 31; x >>= 5; int bh = x;
  int lane = threadIdx.x;
  const __bf16* Ab = q + ((long)bh * N_ + it * 16) * D_;
  const __bf16* Bm = k + ((long)bh * N_ + jt4 * 64) * D_;
  v8f acc[4];
#pragma unroll
  for (int c = 0; c < 4; c++) acc[c] = vzero();
  for (int kk = 0; kk < D_; kk += 32) {
    v16bf a = load_a(Ab + kk, D_, lane);
#pragma unroll
    for (int c = 0; c < 4; c++)
      acc[c] = wmma_bf(a, load_b(Bm + (long)c * 16 * D_ + kk, D_, lane), acc[c]);
  }
  int i0 = it * 16 + ((lane >> 4) << 3);
  int bb = bh >> 3;
#pragma unroll
  for (int c = 0; c < 4; c++) {
    int j = jt4 * 64 + c * 16 + (lane & 15);
#pragma unroll
    for (int v = 0; v < 8; v++) {
      int i = i0 + v;
      long o = ((long)bh << 18) + ((long)i << 9) + j;
      S[o] = acc[c][v] + mask[((long)bb << 18) + ((long)i << 9) + j] + bias[o];
    }
  }
}

// ---------- row softmax over 512, fp32 -> bf16 probs ----------
__global__ void softmax512(const float* __restrict__ S, __bf16* __restrict__ Pr) {
  long row = (long)blockIdx.x * 8 + (threadIdx.x >> 5);
  int lane = threadIdx.x & 31;
  const float* p = S + row * N_;
  float v[16]; float mx = -1e30f;
#pragma unroll
  for (int l = 0; l < 16; l++) { v[l] = p[l * 32 + lane]; mx = fmaxf(mx, v[l]); }
  for (int m = 16; m; m >>= 1) mx = fmaxf(mx, __shfl_xor(mx, m, 32));
  float s = 0.f;
#pragma unroll
  for (int l = 0; l < 16; l++) { v[l] = __expf(v[l] - mx); s += v[l]; }
  for (int m = 16; m; m >>= 1) s += __shfl_xor(s, m, 32);
  float inv = 1.f / s;
#pragma unroll
  for (int l = 0; l < 16; l++) Pr[row * N_ + l * 32 + lane] = f2bf(v[l] * inv);
}

// ---------- O = P @ V (16x96 per wave), fused sigmoid gate: og = sigmoid(g) * O ----------
__global__ __launch_bounds__(32) void attn_av(const __bf16* __restrict__ Pr,
                                              const __bf16* __restrict__ vt,
                                              const __bf16* __restrict__ gsig,
                                              __bf16* __restrict__ og) {
  int x = blockIdx.x; int it = x & 31; x >>= 5; int bh = x;
  int lane = threadIdx.x;
  const __bf16* Ab = Pr + ((long)bh << 18) + (long)it * 16 * N_;
  const __bf16* Bm = vt + (long)bh * D_ * N_;
  v8f acc[6];
#pragma unroll
  for (int c = 0; c < 6; c++) acc[c] = vzero();
  for (int kk = 0; kk < N_; kk += 32) {
    v16bf a = load_a(Ab + kk, N_, lane);
#pragma unroll
    for (int c = 0; c < 6; c++)
      acc[c] = wmma_bf(a, load_b(Bm + (long)c * 16 * N_ + kk, N_, lane), acc[c]);
  }
  int i0 = it * 16 + ((lane >> 4) << 3);
  int b = bh >> 3, hh = bh & 7;
#pragma unroll
  for (int c = 0; c < 6; c++) {
    int d = c * 16 + (lane & 15);
#pragma unroll
    for (int v = 0; v < 8; v++) {
      int i = i0 + v;
      long idx = ((long)(b * N_ + i)) * E_ + hh * D_ + d;
      og[idx] = f2bf(bf2f(gsig[idx]) * acc[c][v]);
    }
  }
}

// ---------- outer product: Ap[b,i,p,e] = sum_d a_o[b,i,d] * w_opm_out[p, d*32+e] ----------
__global__ __launch_bounds__(32) void opm_ap(const __bf16* __restrict__ ab,
                                             const __bf16* __restrict__ W2,
                                             __bf16* __restrict__ Ap) {
  int tile = blockIdx.x; int nt = tile & 127; int mt = tile >> 7;
  int lane = threadIdx.x;
  const __bf16* Ab = ab + (long)mt * 16 * 64;          // a_o = ab[:, 0:32]
  int p = nt >> 1; int e0 = (nt & 1) * 16;
  const __bf16* Bm = W2 + p * 1024 + e0;               // W2[pe, d] strided
  v8f acc = vzero();
  acc = wmma_bf(load_a(Ab, 64, lane), load_b_strided(Bm, 1, 32, lane), acc);
  int n = nt * 16 + (lane & 15); int m0 = mt * 16 + ((lane >> 4) << 3);
#pragma unroll
  for (int v = 0; v < 8; v++) Ap[(long)(m0 + v) * 2048 + n] = f2bf(acc[v]);
}

// ---------- z[b,i,j,p] = sum_e Ap[b,i,p,e] * b_o[b,j,e]; pair += (z+bias)*op_norm ----------
// B fragment shared across 4 consecutive p slices.
__global__ __launch_bounds__(32) void opm_z(const __bf16* __restrict__ Ap,
                                            const __bf16* __restrict__ ab,
                                            const float* __restrict__ pair_in,
                                            const float* __restrict__ b_out,
                                            const float* __restrict__ op_norm,
                                            float* __restrict__ pair_out) {
  int x = blockIdx.x; int jt = x & 31; x >>= 5; int it = x & 31; x >>= 5;
  int p4 = x & 15; x >>= 4; int b = x;
  int lane = threadIdx.x;
  const __bf16* Bm = ab + ((long)b * N_ + jt * 16) * 64 + 32;     // b_o = ab[:, 32:64]
  v16bf bfrag = load_b(Bm, 64, lane);
  float scale = op_norm[0];
  int j = jt * 16 + (lane & 15); int i0 = it * 16 + ((lane >> 4) << 3);
#pragma unroll
  for (int c = 0; c < 4; c++) {
    int p = p4 * 4 + c;
    const __bf16* Ab = Ap + ((long)b * N_ + it * 16) * 2048 + p * 32;
    v8f acc = wmma_bf(load_a(Ab, 2048, lane), bfrag, vzero());
    float bo = b_out[p];
#pragma unroll
    for (int v = 0; v < 8; v++) {
      int i = i0 + v;
      long idx = (((long)(b * N_ + i)) * N_ + j) * P_ + p;
      pair_out[idx] = pair_in[idx] + (acc[v] + bo) * scale;
    }
  }
}

// ---------- triangle a/b: ab = (zp@Wp+bp)*sigmoid(zp@Wg+bg), split into [B,PH,N,N] ----------
__global__ __launch_bounds__(32) void tm_ab(const __bf16* __restrict__ zp,
                                            const __bf16* __restrict__ Wp, const float* __restrict__ bp,
                                            const __bf16* __restrict__ Wg, const float* __restrict__ bg,
                                            __bf16* __restrict__ a_h, __bf16* __restrict__ b_h) {
  int mt = blockIdx.x;
  int lane = threadIdx.x;
  const __bf16* Ab = zp + (long)mt * 16 * 64;
  v8f accp[4], accg[4];
#pragma unroll
  for (int c = 0; c < 4; c++) { accp[c] = vzero(); accg[c] = vzero(); }
  for (int k = 0; k < 64; k += 32) {
    v16bf a = load_a(Ab + k, 64, lane);
#pragma unroll
    for (int c = 0; c < 4; c++) {
      accp[c] = wmma_bf(a, load_b(Wp + c * 16 * 64 + k, 64, lane), accp[c]);
      accg[c] = wmma_bf(a, load_b(Wg + c * 16 * 64 + k, 64, lane), accg[c]);
    }
  }
  int m0 = mt * 16 + ((lane >> 4) << 3);
#pragma unroll
  for (int c = 0; c < 4; c++) {
    int cc = c * 16 + (lane & 15);
    float bpv = bp[cc], bgv = bg[cc];
#pragma unroll
    for (int v = 0; v < 8; v++) {
      int m = m0 + v;
      float val = (accp[c][v] + bpv) * sigmf(accg[c][v] + bgv);
      int b = m >> 18, i = (m >> 9) & 511, j = m & 511;
      int ch = cc & 31;
      __bf16* dst = (cc < 32) ? a_h : b_h;
      dst[(((long)(b * 32 + ch)) * N_ + i) * N_ + j] = f2bf(val);
    }
  }
}

// ---------- 512x512 bf16 transpose via async global->LDS copies (ASYNCcnt path) ----------
__global__ void tr512(const __bf16* __restrict__ in, __bf16* __restrict__ out) {
  __shared__ __bf16 tile[32][40];   // 80B row stride: 16B-aligned rows, padded vs banks
  int x = blockIdx.x;
  int tj = (x & 15) * 32; int ti = ((x >> 4) & 15) * 32;
  long mb = (long)(x >> 8) * ((long)N_ * N_);
  int t = threadIdx.y * 32 + threadIdx.x;       // 0..255
  if (t < 128) {                                 // 128 lanes x 16B = 32x32 bf16 tile
    int r = t >> 2;
    int c = (t & 3) * 8;
    const __bf16* gsrc = in + mb + (long)(ti + r) * N_ + tj + c;
    unsigned lds_off = (unsigned)(size_t)(&tile[r][c]);
    asm volatile("global_load_async_to_lds_b128 %0, %1, off"
                 :: "v"(lds_off), "v"(gsrc) : "memory");
  }
  asm volatile("s_wait_asynccnt 0x0" ::: "memory");
  __syncthreads();
  for (int r = threadIdx.y; r < 32; r += 8)
    out[mb + (long)(tj + r) * N_ + ti + threadIdx.x] = tile[threadIdx.x][r];
}

// ---------- triangle einsums: t[b,h,i,k] = a@b^T + aT@bT^T (16x64 per wave) ----------
__global__ __launch_bounds__(32) void tri(const __bf16* __restrict__ a_h,
                                          const __bf16* __restrict__ b_h,
                                          const __bf16* __restrict__ aT,
                                          const __bf16* __restrict__ bT,
                                          float* __restrict__ t) {
  int x = blockIdx.x; int kt4 = x & 7; x >>= 3; int it = x & 31; x >>= 5;
  int h = x & 31; x >>= 5; int b = x;
  long mo = ((long)(b * 32 + h)) << 18;
  int lane = threadIdx.x;
  const __bf16* A1 = a_h + mo + (long)it * 16 * N_;
  const __bf16* B1 = b_h + mo + (long)kt4 * 64 * N_;
  const __bf16* A2 = aT + mo + (long)it * 16 * N_;
  const __bf16* B2 = bT + mo + (long)kt4 * 64 * N_;
  v8f acc[4];
#pragma unroll
  for (int c = 0; c < 4; c++) acc[c] = vzero();
  for (int kk = 0; kk < N_; kk += 32) {
    v16bf a1 = load_a(A1 + kk, N_, lane);
    v16bf a2 = load_a(A2 + kk, N_, lane);
#pragma unroll
    for (int c = 0; c < 4; c++) {
      acc[c] = wmma_bf(a1, load_b(B1 + (long)c * 16 * N_ + kk, N_, lane), acc[c]);
      acc[c] = wmma_bf(a2, load_b(B2 + (long)c * 16 * N_ + kk, N_, lane), acc[c]);
    }
  }
  int i0 = it * 16 + ((lane >> 4) << 3);
#pragma unroll
  for (int c = 0; c < 4; c++) {
    int k = kt4 * 64 + c * 16 + (lane & 15);
#pragma unroll
    for (int v = 0; v < 8; v++) {
      int i = i0 + v;
      t[(((long)(b * N_ + i)) * N_ + k) * PH_ + h] = acc[c][v];
    }
  }
}

// ---------- LN(t) @ w_tm_z^T + b_tm_z, gated by gpr, accumulated into pair ----------
__global__ __launch_bounds__(32) void tm_out(const float* __restrict__ t,
                                             const float* __restrict__ g, const float* __restrict__ be,
                                             const float* __restrict__ wz, const float* __restrict__ bz,
                                             const __bf16* __restrict__ gpr, float* __restrict__ pair) {
  long row = blockIdx.x; int lane = threadIdx.x;
  float tv = t[row * PH_ + lane];
  float s = tv;
  for (int m = 16; m; m >>= 1) s += __shfl_xor(s, m, 32);
  float mu = s * (1.f / 32.f);
  float d = tv - mu; float q = d * d;
  for (int m = 16; m; m >>= 1) q += __shfl_xor(q, m, 32);
  float inv = rsqrtf(q * (1.f / 32.f) + 1e-5f);
  __shared__ float sh[32];
  sh[lane] = d * inv * g[lane] + be[lane];
  __syncthreads();
#pragma unroll
  for (int half = 0; half < 2; half++) {
    int p = lane + half * 32;
    float dot = bz[p];
    const float* wr = wz + p * 32;
    for (int hh = 0; hh < 32; hh++) dot += wr[hh] * sh[hh];
    long idx = row * P_ + p;
    pair[idx] += bf2f(gpr[idx]) * dot;
  }
}

// ==================================================================================
extern "C" void kernel_launch(void* const* d_in, const int* in_sizes, int n_in,
                              void* d_out, int out_size, void* d_ws, size_t ws_size,
                              hipStream_t stream) {
  (void)in_sizes; (void)n_in; (void)out_size; (void)ws_size;
  const float* x_in   = (const float*)d_in[0];
  const float* pairin = (const float*)d_in[1];
  const float* amask  = (const float*)d_in[2];
  const float* opmask = (const float*)d_in[3];
  const float* opnorm = (const float*)d_in[4];
  // d_in[5] = nlist (unused)
  const float* wq = (const float*)d_in[6],  *wk = (const float*)d_in[7];
  const float* wv = (const float*)d_in[8],  *wg = (const float*)d_in[9];
  const float* bg = (const float*)d_in[10], *wo = (const float*)d_in[11], *bo = (const float*)d_in[12];
  const float* ln_pg = (const float*)d_in[13], *ln_pb = (const float*)d_in[14];
  const float* wb = (const float*)d_in[15], *bb = (const float*)d_in[16];
  const float* ln_ag = (const float*)d_in[17], *ln_ab = (const float*)d_in[18];
  const float* ln_fg = (const float*)d_in[19], *ln_fb = (const float*)d_in[20];
  const float* wffn1 = (const float*)d_in[21], *bffn1 = (const float*)d_in[22];
  const float* wffn2 = (const float*)d_in[23], *bffn2 = (const float*)d_in[24];
  const float* ln_og = (const float*)d_in[25], *ln_ob = (const float*)d_in[26];
  const float* wopm_in = (const float*)d_in[27], *bopm_in = (const float*)d_in[28];
  const float* wopm_out = (const float*)d_in[29], *bopm_out = (const float*)d_in[30];
  const float* ln_tg = (const float*)d_in[31], *ln_tb = (const float*)d_in[32];
  const float* wabp = (const float*)d_in[33], *babp = (const float*)d_in[34];
  const float* wabg = (const float*)d_in[35], *babg = (const float*)d_in[36];
  const float* wtmg = (const float*)d_in[37], *btmg = (const float*)d_in[38];
  const float* wtmz = (const float*)d_in[39], *btmz = (const float*)d_in[40];
  const float* ln_tog = (const float*)d_in[41], *ln_tob = (const float*)d_in[42];
  const float* ln_ffg = (const float*)d_in[43], *ln_ffb = (const float*)d_in[44];
  const float* wpf1 = (const float*)d_in[45], *bpf1 = (const float*)d_in[46];
  const float* wpf2 = (const float*)d_in[47], *bpf2 = (const float*)d_in[48];

  float* out_x = (float*)d_out;
  float* out_pair = out_x + (size_t)BN_ * E_;

  // ---- workspace bump allocator ----
  char* ws = (char*)d_ws; size_t off = 0;
  auto alloc = [&](size_t bytes) -> char* {
    char* p = ws + off; off += (bytes + 255) & ~(size_t)255; return p;
  };
  __bf16* wqb = (__bf16*)alloc((size_t)E_*E_*2);
  __bf16* wkb = (__bf16*)alloc((size_t)E_*E_*2);
  __bf16* wvb = (__bf16*)alloc((size_t)E_*E_*2);
  __bf16* wgb = (__bf16*)alloc((size_t)E_*E_*2);
  __bf16* wob = (__bf16*)alloc((size_t)E_*E_*2);
  __bf16* wf1b = (__bf16*)alloc((size_t)FFN_*E_*2);
  __bf16* wf2b = (__bf16*)alloc((size_t)E_*FFN_*2);
  __bf16* woib = (__bf16*)alloc((size_t)64*E_*2);
  __bf16* woob = (__bf16*)alloc((size_t)64*1024*2);
  __bf16* wabpb = (__bf16*)alloc(64*64*2);
  __bf16* wabgb = (__bf16*)alloc(64*64*2);
  __bf16* wtmgb = (__bf16*)alloc(64*64*2);
  __bf16* wpf1b = (__bf16*)alloc(64*64*2);
  __bf16* wpf2b = (__bf16*)alloc(64*64*2);
  __bf16* hbf  = (__bf16*)alloc((size_t)BN_*E_*2);
  float*  biasb = (float*)alloc((size_t)B_*H_*N_*N_*4);
  __bf16* qb   = (__bf16*)alloc((size_t)BN_*E_*2);
  __bf16* kb   = (__bf16*)alloc((size_t)BN_*E_*2);
  __bf16* vtb  = (__bf16*)alloc((size_t)BN_*E_*2);
  __bf16* gsb  = (__bf16*)alloc((size_t)BN_*E_*2);
  __bf16* ogb  = (__bf16*)alloc((size_t)BN_*E_*2);
  float*  Sb   = (float*)alloc((size_t)B_*H_*N_*N_*4);
  __bf16* prb  = (__bf16*)alloc((size_t)B_*H_*N_*N_*2);
  __bf16* fmid = (__bf16*)alloc((size_t)BN_*FFN_*2);
  __bf16* abb  = (__bf16*)alloc((size_t)BN_*64*2);
  __bf16* Apb  = (__bf16*)alloc((size_t)BN_*2048*2);
  __bf16* zpb  = (__bf16*)alloc((size_t)BNN_*64*2);
  __bf16* gprb = (__bf16*)alloc((size_t)BNN_*64*2);   // reused as pffn mid
  __bf16* ahb  = (__bf16*)alloc((size_t)B_*PH_*N_*N_*2);
  __bf16* bhb  = (__bf16*)alloc((size_t)B_*PH_*N_*N_*2);
  __bf16* aTb  = (__bf16*)alloc((size_t)B_*PH_*N_*N_*2);
  __bf16* bTb  = (__bf16*)alloc((size_t)B_*PH_*N_*N_*2);
  float*  tb   = (float*)alloc((size_t)BNN_*PH_*4);

  auto cvt = [&](const float* src, __bf16* dst, long n) {
    cvt_f2bf<<<(unsigned)((n + 255) / 256), 256, 0, stream>>>(src, dst, n);
  };
  // weight conversions fp32 -> bf16
  cvt(wq, wqb, (long)E_*E_);  cvt(wk, wkb, (long)E_*E_);  cvt(wv, wvb, (long)E_*E_);
  cvt(wg, wgb, (long)E_*E_);  cvt(wo, wob, (long)E_*E_);
  cvt(wffn1, wf1b, (long)FFN_*E_); cvt(wffn2, wf2b, (long)E_*FFN_);
  cvt(wopm_in, woib, 64L*E_); cvt(wopm_out, woob, 64L*1024);
  cvt(wabp, wabpb, 4096); cvt(wabg, wabgb, 4096); cvt(wtmg, wtmgb, 4096);
  cvt(wpf1, wpf1b, 4096); cvt(wpf2, wpf2b, 4096);

  // ===== stage A: attention =====
  ln_rows<<<BN_, 64, 0, stream>>>(x_in, ln_ag, ln_ab, hbf, E_);
  pair_bias<<<BNN_/8, 256, 0, stream>>>(pairin, ln_pg, ln_pb, wb, bb, biasb);

  const unsigned gEE = (BN_/16) * (E_/64);   // 1536
  gemm_qkv<0><<<gEE, 32, 0, stream>>>(hbf, wqb, qb, 0.102062072616f); // 1/sqrt(96)
  gemm_qkv<1><<<gEE, 32, 0, stream>>>(hbf, wkb, kb, 1.f);
  gemm_qkv<2><<<gEE, 32, 0, stream>>>(hbf, wvb, vtb, 1.f);
  gemm_rm<ACT_SIG, false><<<gEE, 32, 0, stream>>>(hbf, E_, wgb, E_, bg, gsb, nullptr,
                                                  nullptr, nullptr, E_, E_, E_);
  attn_logits<<<B_*H_*32*8, 32, 0, stream>>>(qb, kb, amask, biasb, Sb);
  softmax512<<<B_*H_*N_/8, 256, 0, stream>>>(Sb, prb);
  attn_av<<<B_*H_*32, 32, 0, stream>>>(prb, vtb, gsb, ogb);
  gemm_rm<ACT_NONE, true><<<gEE, 32, 0, stream>>>(ogb, E_, wob, E_, bo, nullptr, out_x,
                                                  x_in, nullptr, E_, E_, E_);
  // ===== stage B: token FFN =====
  ln_rows<<<BN_, 64, 0, stream>>>(out_x, ln_fg, ln_fb, hbf, E_);
  gemm_rm<ACT_GELU, false><<<(BN_/16)*(FFN_/64), 32, 0, stream>>>(hbf, E_, wf1b, E_, bffn1,
                                                  fmid, nullptr, nullptr, nullptr, FFN_, E_, FFN_);
  gemm_rm<ACT_NONE, true><<<gEE, 32, 0, stream>>>(fmid, FFN_, wf2b, FFN_, bffn2, nullptr,
                                                  out_x, out_x, nullptr, E_, FFN_, E_);
  // ===== stage C: outer product pair update =====
  ln_rows<<<BN_, 64, 0, stream>>>(out_x, ln_og, ln_ob, hbf, E_);
  gemm_rm<ACT_NONE, false><<<(BN_/16), 32, 0, stream>>>(hbf, E_, woib, E_, bopm_in,
                                                  abb, nullptr, nullptr, opmask, 64, E_, 64);
  opm_ap<<<(BN_/16)*128, 32, 0, stream>>>(abb, woob, Apb);
  opm_z<<<B_*16*32*32, 32, 0, stream>>>(Apb, abb, pairin, bopm_out, opnorm, out_pair);
  // ===== stage D: triangle multiplication =====
  ln_rows<<<BNN_, 64, 0, stream>>>(out_pair, ln_tg, ln_tb, zpb, 64);
  gemm_rm<ACT_NONE, false><<<(BNN_/16), 32, 0, stream>>>(zpb, 64, wtmgb, 64, btmg,
                                                  gprb, nullptr, nullptr, nullptr, 64, 64, 64);
  tm_ab<<<(BNN_/16), 32, 0, stream>>>(zpb, wabpb, babp, wabgb, babg, ahb, bhb);
  tr512<<<B_*PH_*256, dim3(32, 8), 0, stream>>>(ahb, aTb);
  tr512<<<B_*PH_*256, dim3(32, 8), 0, stream>>>(bhb, bTb);
  tri<<<B_*PH_*32*8, 32, 0, stream>>>(ahb, bhb, aTb, bTb, tb);
  tm_out<<<BNN_, 32, 0, stream>>>(tb, ln_tog, ln_tob, wtmz, btmz, gprb, out_pair);
  // ===== stage E: pair FFN =====
  ln_rows<<<BNN_, 64, 0, stream>>>(out_pair, ln_ffg, ln_ffb, zpb, 64);
  gemm_rm<ACT_GELU, false><<<(BNN_/16), 32, 0, stream>>>(zpb, 64, wpf1b, 64, bpf1,
                                                  gprb, nullptr, nullptr, nullptr, 64, 64, 64);
  gemm_rm<ACT_NONE, true><<<(BNN_/16), 32, 0, stream>>>(gprb, 64, wpf2b, 64, bpf2, nullptr,
                                                  out_pair, out_pair, nullptr, 64, 64, 64);
}
